// LSTM_70866960384249
// MI455X (gfx1250) — matrix operand order repaired and checked by reading, not compiled, not run on hardware
//
#include <hip/hip_runtime.h>

// Problem constants (match reference)
#define Bc 128
#define Tc 512
#define Dc 256
#define Hc 512
#define BHc (Bc * Hc)   // 65536

typedef __attribute__((ext_vector_type(16))) __bf16 v16bf;
typedef __attribute__((ext_vector_type(8)))  __bf16 v8bf;
typedef __attribute__((ext_vector_type(8)))  float  v8f;

__device__ __forceinline__ v16bf make16(v8bf lo, v8bf hi) {
    return __builtin_shufflevector(lo, hi, 0,1,2,3,4,5,6,7,8,9,10,11,12,13,14,15);
}

// ---- WMMA fragment loaders (CDNA5 wave32 layouts, CDNA5 ISA §7.12.2) ----
// A matrix 16x32 bf16: lane L holds row M=L&15; lanes<16 get K runs [kb..kb+7] with kb=0,
// lanes>=16 kb=8; second 8 halfs are K+16. So per lane: two contiguous 16B loads.
__device__ __forceinline__ v16bf load_a_bf16(const __bf16* __restrict__ rowp, int k0, int lane) {
    const int kb = (lane & 16) ? 8 : 0;
    const __bf16* p = rowp + k0 + kb;
    v8bf lo = *(const v8bf*)(p);
    v8bf hi = *(const v8bf*)(p + 16);
    return make16(lo, hi);
}

// Same A layout but source is fp32 (convert on the fly to bf16).
__device__ __forceinline__ v16bf load_a_f32(const float* __restrict__ rowp, int k0, int lane) {
    const int kb = (lane & 16) ? 8 : 0;
    const float* p = rowp + k0 + kb;
    v16bf a;
#pragma unroll
    for (int i = 0; i < 8; ++i) {
        a[i]     = (__bf16)p[i];
        a[8 + i] = (__bf16)p[16 + i];
    }
    return a;
}

// B matrix 32x16 bf16 with B[k][n] = W[n][k] (W row-major [4H][ldk]):
// lane L holds column N=L&15; lanes<16 K=[k0..k0+7]+[k0+16..k0+23], lanes>=16 K=[k0+8..k0+15]+[k0+24..k0+31]
// -> two contiguous 16B runs per lane along W's row.
__device__ __forceinline__ v16bf load_b_w(const __bf16* __restrict__ W, int ldk,
                                          int n0, int k0, int lane) {
    const int n  = n0 + (lane & 15);
    const int kb = (lane & 16) ? 8 : 0;
    const __bf16* p = W + (size_t)n * ldk + k0 + kb;
    v8bf lo = *(const v8bf*)(p);
    v8bf hi = *(const v8bf*)(p + 16);
    return make16(lo, hi);
}

__device__ __forceinline__ float fsigmoid(float x) { return 1.f / (1.f + __expf(-x)); }
__device__ __forceinline__ float ftanh_(float x)   { return 2.f / (1.f + __expf(-2.f * x)) - 1.f; }

// ---- One LSTM timestep: gates = [x_t | h] @ [W_ih | W_hh]^T + bias, then cell update ----
// Block = 4 waves (128 thr); wave g computes gate g's 16x16 tile for (m0, j0).
// Grid = (B/16, H/16) = (8, 32). h double-buffered across steps (no cross-block race);
// c owned exclusively per tile.
template<int DIN, bool XF32>
__global__ __launch_bounds__(128) void lstm_step_kernel(
    const float*  __restrict__ xin_f,   // layer0 input: x [B,T,DIN] fp32 (XF32)
    const __bf16* __restrict__ xin_b,   // layer1 input: out0 [B,T,DIN] bf16 (!XF32)
    const __bf16* __restrict__ Wih,     // [4H, DIN] bf16
    const __bf16* __restrict__ Whh,     // [4H, H]   bf16
    const float*  __restrict__ bias,    // [4H] (b_ih + b_hh)
    const __bf16* __restrict__ h_read,  // [B*H] bf16, parity t%2
    __bf16*       __restrict__ h_write, // [B*H] bf16, parity (t+1)%2
    float*        __restrict__ c_ws,    // [B*H] fp32, in-place
    float*        __restrict__ out_f,   // layer1: d_out out1 region (else null)
    __bf16*       __restrict__ out_b,   // layer0: out0 bf16 buffer (else null)
    float*        __restrict__ hn_out,  // d_out h_n region for this layer
    float*        __restrict__ cn_out,  // d_out c_n region for this layer
    int t, int is_last)
{
    const int lane = threadIdx.x & 31;
    const int g    = threadIdx.x >> 5;        // gate id: 0=i 1=f 2=g 3=o (PyTorch order)
    const int m0   = blockIdx.x * 16;         // batch tile origin
    const int j0   = blockIdx.y * 16;         // hidden tile origin
    const int n0   = g * Hc + j0;             // column base within [4H]
    const int m    = m0 + (lane & 15);        // A-fragment row for this lane

    v8f acc = {};

    // Input-to-gate contribution (K = DIN)
    if constexpr (XF32) {
        const float* rowp = xin_f + ((size_t)m * Tc + t) * DIN;
#pragma unroll 4
        for (int k0 = 0; k0 < DIN; k0 += 32) {
            v16bf a = load_a_f32(rowp, k0, lane);
            v16bf b = load_b_w(Wih, DIN, n0, k0, lane);
            acc = __builtin_amdgcn_wmma_f32_16x16x32_bf16(false, a, false, b,
                                                          (short)0, acc, false, false);
        }
    } else {
        const __bf16* rowp = xin_b + ((size_t)m * Tc + t) * DIN;
#pragma unroll 4
        for (int k0 = 0; k0 < DIN; k0 += 32) {
            v16bf a = load_a_bf16(rowp, k0, lane);
            v16bf b = load_b_w(Wih, DIN, n0, k0, lane);
            acc = __builtin_amdgcn_wmma_f32_16x16x32_bf16(false, a, false, b,
                                                          (short)0, acc, false, false);
        }
    }

    // Hidden-to-gate contribution (K = H)
    {
        const __bf16* rowp = h_read + (size_t)m * Hc;
#pragma unroll 4
        for (int k0 = 0; k0 < Hc; k0 += 32) {
            v16bf a = load_a_bf16(rowp, k0, lane);
            v16bf b = load_b_w(Whh, Hc, n0, k0, lane);
            acc = __builtin_amdgcn_wmma_f32_16x16x32_bf16(false, a, false, b,
                                                          (short)0, acc, false, false);
        }
    }

    // Exchange the 4 gate tiles through LDS. C/D layout: VGPR r -> row r (+8 if lane>=16),
    // col = lane&15.
    __shared__ float gsm[4][16][17];
    const int col   = lane & 15;
    const int rbase = (lane & 16) ? 8 : 0;
    const float bv  = bias[n0 + col];
#pragma unroll
    for (int r = 0; r < 8; ++r) gsm[g][rbase + r][col] = acc[r] + bv;
    __syncthreads();

    // Cell update: 256 elements / 128 threads
    for (int e = threadIdx.x; e < 256; e += 128) {
        const int row = e >> 4, cc = e & 15;
        const float gi = gsm[0][row][cc];
        const float gf = gsm[1][row][cc];
        const float gg = gsm[2][row][cc];
        const float go = gsm[3][row][cc];
        const float iv = fsigmoid(gi);
        const float fv = fsigmoid(gf);
        const float gv = ftanh_(gg);
        const float ov = fsigmoid(go);
        const int b  = m0 + row;
        const int hi = j0 + cc;
        const size_t ci = (size_t)b * Hc + hi;
        const float c = fv * c_ws[ci] + iv * gv;
        c_ws[ci] = c;
        const float h = ov * ftanh_(c);
        h_write[ci] = (__bf16)h;
        const size_t oi = ((size_t)b * Tc + t) * Hc + hi;
        if (out_f) out_f[oi] = h;
        if (out_b) out_b[oi] = (__bf16)h;
        if (is_last) { hn_out[ci] = h; cn_out[ci] = c; }
    }
}

// ---- Setup: convert weights to bf16 (L2-resident across all 1024 steps), fuse biases,
// init h (bf16, parity 0) and c (fp32). ----
__global__ void lstm_setup_kernel(
    const float* __restrict__ Wih0, const float* __restrict__ Whh0,
    const float* __restrict__ Wih1, const float* __restrict__ Whh1,
    const float* __restrict__ bih0, const float* __restrict__ bhh0,
    const float* __restrict__ bih1, const float* __restrict__ bhh1,
    const float* __restrict__ h0,   const float* __restrict__ c0,
    __bf16* __restrict__ wih0b, __bf16* __restrict__ whh0b,
    __bf16* __restrict__ wih1b, __bf16* __restrict__ whh1b,
    float* __restrict__ bias0, float* __restrict__ bias1,
    __bf16* __restrict__ hbf /* [L][2][B*H] */, float* __restrict__ cws /* [L][B*H] */)
{
    const int N = 4 * Hc * Hc;  // largest buffer: 1,048,576
    for (int i = blockIdx.x * blockDim.x + threadIdx.x; i < N;
         i += gridDim.x * blockDim.x) {
        if (i < 4 * Hc * Dc) wih0b[i] = (__bf16)Wih0[i];
        whh0b[i] = (__bf16)Whh0[i];
        wih1b[i] = (__bf16)Wih1[i];
        whh1b[i] = (__bf16)Whh1[i];
        if (i < 4 * Hc) {
            bias0[i] = bih0[i] + bhh0[i];
            bias1[i] = bih1[i] + bhh1[i];
        }
        if (i < 2 * BHc) {
            const int l = i / BHc, j = i - l * BHc;
            hbf[(size_t)l * 2 * BHc + j] = (__bf16)h0[i];  // parity-0 buffer
            cws[i] = c0[i];
        }
    }
}

// ---- y_pred = out1[:, T-1, :] @ W_lin^T + b_lin ; one wave per batch row ----
__global__ void ypred_kernel(const float* __restrict__ out1,
                             const float* __restrict__ Wlin,
                             const float* __restrict__ blin,
                             float* __restrict__ y)
{
    const int b = blockIdx.x;
    const int lane = threadIdx.x;
    const float* row = out1 + ((size_t)b * Tc + (Tc - 1)) * Hc;
    float s = 0.f;
    for (int k = lane; k < Hc; k += 32) s += row[k] * Wlin[k];
#pragma unroll
    for (int off = 16; off > 0; off >>= 1) s += __shfl_xor(s, off, 32);
    if (lane == 0) y[b] = s + blin[0];
}

extern "C" void kernel_launch(void* const* d_in, const int* in_sizes, int n_in,
                              void* d_out, int out_size, void* d_ws, size_t ws_size,
                              hipStream_t stream) {
    (void)in_sizes; (void)n_in; (void)out_size; (void)ws_size;

    const float* x    = (const float*)d_in[0];
    const float* h0   = (const float*)d_in[1];
    const float* c0   = (const float*)d_in[2];
    const float* Wih0 = (const float*)d_in[3];
    const float* Whh0 = (const float*)d_in[4];
    const float* bih0 = (const float*)d_in[5];
    const float* bhh0 = (const float*)d_in[6];
    const float* Wih1 = (const float*)d_in[7];
    const float* Whh1 = (const float*)d_in[8];
    const float* bih1 = (const float*)d_in[9];
    const float* bhh1 = (const float*)d_in[10];
    const float* Wlin = (const float*)d_in[11];
    const float* blin = (const float*)d_in[12];
    float* out = (float*)d_out;

    // Output layout (flattened return order): out1 [B,T,H], y_pred [B], h_n [2,B,H], c_n [2,B,H]
    const size_t O_Y  = (size_t)Bc * Tc * Hc;        // 33,554,432
    const size_t O_HN = O_Y + Bc;
    const size_t O_CN = O_HN + 2 * (size_t)BHc;

    // Workspace carve-out (~75 MB total)
    char* p = (char*)d_ws;
    auto carve = [&](size_t bytes) {
        char* r = p;
        p += (bytes + 255) & ~(size_t)255;
        return r;
    };
    __bf16* wih0b = (__bf16*)carve((size_t)4 * Hc * Dc * 2);   // 1 MB
    __bf16* whh0b = (__bf16*)carve((size_t)4 * Hc * Hc * 2);   // 2 MB
    __bf16* wih1b = (__bf16*)carve((size_t)4 * Hc * Hc * 2);   // 2 MB
    __bf16* whh1b = (__bf16*)carve((size_t)4 * Hc * Hc * 2);   // 2 MB
    float*  bias0 = (float*)carve((size_t)4 * Hc * 4);
    float*  bias1 = (float*)carve((size_t)4 * Hc * 4);
    __bf16* hbf   = (__bf16*)carve((size_t)2 * 2 * BHc * 2);   // [L][parity][B*H]
    float*  cws   = (float*)carve((size_t)2 * BHc * 4);        // [L][B*H]
    __bf16* out0b = (__bf16*)carve((size_t)Bc * Tc * Hc * 2);  // 64 MB, layer0 output bf16

    lstm_setup_kernel<<<4096, 256, 0, stream>>>(
        Wih0, Whh0, Wih1, Whh1, bih0, bhh0, bih1, bhh1, h0, c0,
        wih0b, whh0b, wih1b, whh1b, bias0, bias1, hbf, cws);

    const dim3 sgrid(Bc / 16, Hc / 16);  // (8, 32)

    // Layer 0: input = x (fp32, converted on load), K = D + H
    for (int t = 0; t < Tc; ++t) {
        lstm_step_kernel<Dc, true><<<sgrid, 128, 0, stream>>>(
            x, nullptr, wih0b, whh0b, bias0,
            hbf + (size_t)(0 * 2 + (t & 1)) * BHc,
            hbf + (size_t)(0 * 2 + ((t + 1) & 1)) * BHc,
            cws + 0,
            nullptr, out0b,
            out + O_HN + 0, out + O_CN + 0,
            t, t == Tc - 1);
    }

    // Layer 1: input = out0 (bf16), K = H + H; writes out1 fp32 straight to d_out
    for (int t = 0; t < Tc; ++t) {
        lstm_step_kernel<Hc, false><<<sgrid, 128, 0, stream>>>(
            nullptr, out0b, wih1b, whh1b, bias1,
            hbf + (size_t)(2 + (t & 1)) * BHc,
            hbf + (size_t)(2 + ((t + 1) & 1)) * BHc,
            cws + BHc,
            out, nullptr,
            out + O_HN + BHc, out + O_CN + BHc,
            t, t == Tc - 1);
    }

    ypred_kernel<<<Bc, 32, 0, stream>>>(out, Wlin, blin, out + O_Y);
}
